// STAttnGraphConv_22943715295394
// MI455X (gfx1250) — matrix operand order repaired and checked by reading, not compiled, not run on hardware
//
#include <hip/hip_runtime.h>
#include <hip/hip_bf16.h>

// ---------------------------------------------------------------------------
// STAttnGraphConv for MI455X (gfx1250, wave32, WMMA).
// Shapes: b=16, n=2048, t=12, k=32, d_q=32, d_k=64, d_c=10, n_head=4, d_out=32
// Dominant cost: two rounds of z = adj@z + xp@W  (~103 GFLOP) -> bf16 WMMA.
// adj is NEVER materialized in HBM: per 16-row tile we build exp(relu(.)) in
// LDS (bf16) and fold the softmax denominator into the accumulator post-hoc.
// ---------------------------------------------------------------------------

#define NB   16
#define NN   2048
#define NT   12
#define NK   32
#define DQ   32
#define DK   64
#define DC   10
#define NH   4
#define DOUT 32

typedef __bf16  v16bf __attribute__((ext_vector_type(16)));
typedef float   v8f   __attribute__((ext_vector_type(8)));

union BfPack { v16bf v; unsigned int u[8]; };

__device__ __forceinline__ __bf16 f2bf(float f) {
    union { float f; unsigned int u; } c; c.f = f;
    unsigned int r = (c.u + 0x7FFFu + ((c.u >> 16) & 1u)) >> 16;   // RNE
    union { unsigned short s; __bf16 b; } o; o.s = (unsigned short)r;
    return o.b;
}

// ---------------- front-end reductions (linearity: mean(x)@W == mean(x@W)) --

__global__ void reduce_t_kernel(const float* __restrict__ x, float* __restrict__ xbar_t) {
    // xbar_t[b,n,k] = mean_t x[b,n,t,k]   (16*2048*32 threads)
    unsigned idx = blockIdx.x * blockDim.x + threadIdx.x;
    if (idx >= NB * NN * NK) return;
    unsigned k  = idx & 31u;
    unsigned bn = idx >> 5;
    const float* xr = x + (size_t)bn * NT * NK + k;
    float s = 0.f;
    #pragma unroll
    for (int t = 0; t < NT; ++t) s += xr[t * NK];
    xbar_t[idx] = s * (1.0f / NT);
}

__global__ void reduce_n_kernel(const float* __restrict__ x, float* __restrict__ xbar_n) {
    // xbar_n[b,t,k] = mean_n x[b,n,t,k]   (16*12*32 = 6144 threads)
    unsigned idx = blockIdx.x * blockDim.x + threadIdx.x;
    if (idx >= NB * NT * NK) return;
    unsigned k = idx & 31u;
    unsigned t = (idx >> 5) % NT;
    unsigned b = idx / (NT * NK);
    float s = 0.f;
    for (int n = 0; n < NN; ++n)
        s += x[(((size_t)b * NN + n) * NT + t) * NK + k];
    xbar_n[idx] = s * (1.0f / NN);
}

__global__ void q0_kernel(const float* __restrict__ xbar_t, const float* __restrict__ W_Q0,
                          float* __restrict__ q0) {
    // q0[b,n,dq] = xbar_t[b,n,:] @ W_Q0
    unsigned idx = blockIdx.x * blockDim.x + threadIdx.x;
    if (idx >= NB * NN * DQ) return;
    unsigned dq = idx & 31u;
    unsigned bn = idx >> 5;
    float s = 0.f;
    #pragma unroll
    for (int k = 0; k < NK; ++k) s += xbar_t[(size_t)bn * NK + k] * W_Q0[k * DQ + dq];
    q0[idx] = s;
}

__global__ void qt_kernel(const float* __restrict__ xbar_n, const float* __restrict__ W_Q0,
                          const float* __restrict__ W_QT, float* __restrict__ qt) {
    // per-b block: p[t,dq] = xbar_n@W_Q0 ; qt[b,:] = p_flat @ W_QT  [384x64]
    int b = blockIdx.x;
    __shared__ float p[NT * DQ];
    for (int e = threadIdx.x; e < NT * DQ; e += blockDim.x) {
        int t = e >> 5, dq = e & 31;
        float s = 0.f;
        #pragma unroll
        for (int k = 0; k < NK; ++k) s += xbar_n[(b * NT + t) * NK + k] * W_Q0[k * DQ + dq];
        p[e] = s;
    }
    __syncthreads();
    if (threadIdx.x < DK) {
        float s = 0.f;
        for (int i = 0; i < NT * DQ; ++i) s += p[i] * W_QT[i * DK + threadIdx.x];
        qt[b * DK + threadIdx.x] = s;
    }
}

__global__ void qs_kernel(const float* __restrict__ q0, const float* __restrict__ W_QS,
                          float* __restrict__ qs) {
    // block = (b, j): qs[b,j] = sum_i q0_flat[b,i] * W_QS[i,j], i < 65536
    int b = blockIdx.x >> 6, j = blockIdx.x & 63;
    __shared__ float red[256];
    float s = 0.f;
    for (int i = threadIdx.x; i < NN * DQ; i += 256)
        s += q0[(size_t)b * NN * DQ + i] * W_QS[(size_t)i * DK + j];
    red[threadIdx.x] = s;
    __syncthreads();
    for (int o = 128; o > 0; o >>= 1) {
        if ((int)threadIdx.x < o) red[threadIdx.x] += red[threadIdx.x + o];
        __syncthreads();
    }
    if (threadIdx.x == 0) qs[b * DK + j] = red[0];
}

__global__ void attn_kernel(const float* __restrict__ qt, const float* __restrict__ qs,
                            const float* __restrict__ K_T, const float* __restrict__ K_S,
                            float* __restrict__ attn_t, float* __restrict__ attn_s) {
    int b = blockIdx.x;
    if (threadIdx.x != 0) return;
    const float scale = 0.125f;  // 1/sqrt(64)
    float lt[NH], ls[NH];
    #pragma unroll
    for (int h = 0; h < NH; ++h) {
        float st = 0.f, ss = 0.f;
        for (int j = 0; j < DK; ++j) {
            st += qt[b * DK + j] * K_T[h * DK + j];
            ss += qs[b * DK + j] * K_S[h * DK + j];
        }
        lt[h] = st * scale; ls[h] = ss * scale;
    }
    float mt = lt[0], ms = ls[0];
    #pragma unroll
    for (int h = 1; h < NH; ++h) { mt = fmaxf(mt, lt[h]); ms = fmaxf(ms, ls[h]); }
    float dt = 0.f, dsm = 0.f;
    #pragma unroll
    for (int h = 0; h < NH; ++h) {
        lt[h] = __expf(lt[h] - mt); ls[h] = __expf(ls[h] - ms);
        dt += lt[h]; dsm += ls[h];
    }
    #pragma unroll
    for (int h = 0; h < NH; ++h) {
        attn_t[b * NH + h] = lt[h] / dt;
        attn_s[b * NH + h] = ls[h] / dsm;
    }
}

__global__ void ets_kernel(const float* __restrict__ V_T, const float* __restrict__ V_S,
                           const float* __restrict__ at, const float* __restrict__ as,
                           float* __restrict__ ets) {
    unsigned idx = blockIdx.x * blockDim.x + threadIdx.x;   // 16*2048*10
    if (idx >= NB * NN * DC) return;
    unsigned c = idx % DC;
    unsigned n = (idx / DC) % NN;
    unsigned b = idx / (NN * DC);
    float s = 0.f;
    #pragma unroll
    for (int h = 0; h < NH; ++h)
        s += at[b * NH + h] * V_T[((size_t)h * NN + n) * DC + c]
           + as[b * NH + h] * V_S[((size_t)h * NN + n) * DC + c];
    ets[idx] = s;
}

// ---------------- z0 = xp @ W2, stored bf16 in [b][t][d][m] (m fastest) -----

__global__ void z0_kernel(const float* __restrict__ x, const float* __restrict__ W2,
                          __bf16* __restrict__ z0) {
    size_t idx = (size_t)blockIdx.x * blockDim.x + threadIdx.x;  // 16*12*32*2048
    unsigned m = idx & 2047u;
    unsigned d = (idx >> 11) & 31u;
    unsigned rest = (unsigned)(idx >> 16);   // b*12 + t
    unsigned t = rest % NT, b = rest / NT;
    if (b >= NB) return;
    const float* xr = x + (((size_t)b * NN + m) * NT + t) * NK;
    float s = 0.f;
    #pragma unroll
    for (int k = 0; k < NK; ++k) s += xr[k] * W2[k * DOUT + d];
    z0[idx] = f2bf(s);
}

// ---------------- fused adj-softmax + graph conv step (WMMA core) -----------
// One block = (16-node row tile, b). 384 threads = 12 waves; wave w owns t=w.
// z_next[b,t,i,d] = (1/denom_i) * sum_m e[i,m]*zin[b,t,m,d] + xp[b,t,i,:]@W

__global__ __launch_bounds__(384)
void conv_kernel(const float* __restrict__ ets,     // [16][2048][10]
                 const float* __restrict__ dis,     // [2048][2048]
                 const __bf16* __restrict__ zin,    // [16][12][32][2048] bf16
                 const float* __restrict__ x,       // [16][2048][12][32]
                 const float* __restrict__ W,       // [32][32]
                 void* __restrict__ outp, int final_step) {
    __shared__ __bf16 e_tile[16][NN];      // 64 KB: exp(relu(dis+ets.ets)) bf16
    __shared__ float  ets_i[16][12];       // row ets vectors (10, padded)
    __shared__ float  denom[16];
    __shared__ float  inv_den[16];
    __shared__ float  xp_st[NT][16][NK];   // 24 KB: per-t xp tile
    __shared__ float  w_st[NK][DOUT];      // 4 KB

    const int tile = blockIdx.x;           // 0..127
    const int b    = blockIdx.y;           // 0..15
    const int i0   = tile * 16;
    const int tid  = threadIdx.x;          // 0..383
    const int lane = tid & 31;
    const int wid  = tid >> 5;             // t index, 0..11
    const int hi   = lane >> 4;            // half-wave select
    const int lrow = lane & 15;

    // ---- phase 0: stage tiles
    if (tid < 16 * DC) { int i = tid / DC, c = tid % DC;
        ets_i[i][c] = ets[((size_t)b * NN + i0 + i) * DC + c]; }
    if (tid < 16) denom[tid] = 0.f;
    for (int e = tid; e < NK * DOUT; e += 384) w_st[e >> 5][e & 31] = W[e];
    {
        const int t = wid;
        for (int e = lane; e < 16 * NK; e += 32) {
            int i = e >> 5, k = e & 31;
            xp_st[t][i][k] = x[(((size_t)b * NN + i0 + i) * NT + t) * NK + k];
        }
    }
    __syncthreads();

    // ---- phase 1: e tile + row sums (softmax denominator)
    float pacc[16];
    #pragma unroll
    for (int i = 0; i < 16; ++i) pacc[i] = 0.f;
    for (int m = tid; m < NN; m += 384) {
        float em[DC];
        #pragma unroll
        for (int c = 0; c < DC; ++c) em[c] = ets[((size_t)b * NN + m) * DC + c];
        #pragma unroll
        for (int i = 0; i < 16; ++i) {
            float v = dis[(size_t)(i0 + i) * NN + m];
            #pragma unroll
            for (int c = 0; c < DC; ++c) v += ets_i[i][c] * em[c];
            float ev = __expf(fmaxf(v, 0.f));
            e_tile[i][m] = f2bf(ev);
            pacc[i] += ev;
        }
    }
    #pragma unroll
    for (int i = 0; i < 16; ++i) atomicAdd(&denom[i], pacc[i]);
    __syncthreads();
    if (tid < 16) inv_den[tid] = 1.0f / denom[tid];
    __syncthreads();

    // ---- phase 2: WMMA. A = e tile rows [16 x 32-chunk], B = zin [32 x 16].
    const int t = wid;
    v8f acc0 = {}; v8f acc1 = {};

    // bias = xp_tile @ W via one bf16 WMMA per N-tile
    BfPack axp;
    #pragma unroll
    for (int v = 0; v < 8; ++v) {
        // 16-bit A 16x32 layout: lanes<16 hold K{0..7,16..23}, lanes>=16 K{8..15,24..31}
        int kk = (v < 4) ? (hi * 8 + 2 * v) : (16 + hi * 8 + 2 * (v - 4));
        axp.v[2 * v]     = f2bf(xp_st[t][lrow][kk]);
        axp.v[2 * v + 1] = f2bf(xp_st[t][lrow][kk + 1]);
    }
    v8f bias0, bias1;
    #pragma unroll
    for (int nt = 0; nt < 2; ++nt) {
        BfPack bw;
        int col = lrow + 16 * nt;
        #pragma unroll
        for (int v = 0; v < 8; ++v) {
            // 16-bit B 32x16 layout: lanes<16 K{0..7,16..23}, lanes>=16 K{8..15,24..31}, N=lane&15
            int kk = (v < 4) ? (hi * 8 + 2 * v) : (16 + hi * 8 + 2 * (v - 4));
            bw.v[2 * v]     = f2bf(w_st[kk][col]);
            bw.v[2 * v + 1] = f2bf(w_st[kk + 1][col]);
        }
        v8f zc = {};
        v8f r = __builtin_amdgcn_wmma_f32_16x16x32_bf16(false, axp.v, false, bw.v,
                                                        (short)0, zc, false, false);
        if (nt == 0) bias0 = r; else bias1 = r;
    }

    // main K-loop over m in chunks of 32
    const __bf16* zbase = zin + ((size_t)b * NT + t) * DOUT * NN;  // [d][m]
    for (int mc = 0; mc < NN; mc += 32) {
        BfPack a;
        #pragma unroll
        for (int v = 0; v < 8; ++v) {
            int kk = mc + ((v < 4) ? (hi * 8 + 2 * v) : (16 + hi * 8 + 2 * (v - 4)));
            a.u[v] = *(const unsigned int*)&e_tile[lrow][kk];     // bf16 pair
        }
        #pragma unroll
        for (int nt = 0; nt < 2; ++nt) {
            int col = lrow + 16 * nt;
            BfPack bb;
            #pragma unroll
            for (int v = 0; v < 8; ++v) {
                int kk = (v < 4) ? (hi * 8 + 2 * v) : (16 + hi * 8 + 2 * (v - 4));
                bb.u[v] = *(const unsigned int*)(zbase + (size_t)col * NN + mc + kk);
            }
            if (nt == 0)
                acc0 = __builtin_amdgcn_wmma_f32_16x16x32_bf16(false, a.v, false, bb.v,
                                                               (short)0, acc0, false, false);
            else
                acc1 = __builtin_amdgcn_wmma_f32_16x16x32_bf16(false, a.v, false, bb.v,
                                                               (short)0, acc1, false, false);
        }
    }

    // ---- writeback: scale by 1/denom (softmax) + bias.
    // C/D layout: VGPR r -> (M = r + 8*hi, N = lane&15), per N-tile col += 16.
    #pragma unroll
    for (int nt = 0; nt < 2; ++nt) {
        int col = lrow + 16 * nt;
        #pragma unroll
        for (int r = 0; r < 8; ++r) {
            int M = r + 8 * hi;
            float a  = (nt == 0) ? acc0[r] : acc1[r];
            float bi = (nt == 0) ? bias0[r] : bias1[r];
            float val = a * inv_den[M] + bi;
            int i = i0 + M;
            if (final_step) {
                // d_out: [b][n][t][d] fp32
                ((float*)outp)[(((size_t)b * NN + i) * NT + t) * DOUT + col] = val;
            } else {
                // z1: [b][t][d][m] bf16
                ((__bf16*)outp)[((size_t)b * NT + t) * DOUT * NN + (size_t)col * NN + i] = f2bf(val);
            }
        }
    }
}

// ---------------------------------------------------------------------------

extern "C" void kernel_launch(void* const* d_in, const int* in_sizes, int n_in,
                              void* d_out, int out_size, void* d_ws, size_t ws_size,
                              hipStream_t stream) {
    const float* x     = (const float*)d_in[0];
    const float* dis   = (const float*)d_in[1];
    const float* K_S   = (const float*)d_in[2];
    const float* V_S   = (const float*)d_in[3];
    const float* K_T   = (const float*)d_in[4];
    const float* V_T   = (const float*)d_in[5];
    const float* W_Q0  = (const float*)d_in[6];
    const float* W_QS  = (const float*)d_in[7];
    const float* W_QT  = (const float*)d_in[8];
    const float* W2    = (const float*)d_in[9];
    const float* W_s0  = (const float*)d_in[10];
    const float* W_s1  = (const float*)d_in[11];

    char* ws = (char*)d_ws;
    float* xbar_t = (float*)ws;  ws += (size_t)NB * NN * NK * 4;        // 4 MB
    float* xbar_n = (float*)ws;  ws += (size_t)NB * NT * NK * 4;
    float* q0     = (float*)ws;  ws += (size_t)NB * NN * DQ * 4;        // 4 MB
    float* qt     = (float*)ws;  ws += (size_t)NB * DK * 4;
    float* qs     = (float*)ws;  ws += (size_t)NB * DK * 4;
    float* attn_t = (float*)ws;  ws += (size_t)NB * NH * 4;
    float* attn_s = (float*)ws;  ws += (size_t)NB * NH * 4;
    float* etsb   = (float*)ws;  ws += (size_t)NB * NN * DC * 4;        // 1.3 MB
    __bf16* z0    = (__bf16*)ws; ws += (size_t)NB * NT * DOUT * NN * 2; // 25 MB
    __bf16* z1    = (__bf16*)ws; ws += (size_t)NB * NT * DOUT * NN * 2; // 25 MB

    reduce_t_kernel<<<(NB * NN * NK) / 256, 256, 0, stream>>>(x, xbar_t);
    reduce_n_kernel<<<(NB * NT * NK + 255) / 256, 256, 0, stream>>>(x, xbar_n);
    q0_kernel<<<(NB * NN * DQ) / 256, 256, 0, stream>>>(xbar_t, W_Q0, q0);
    qt_kernel<<<NB, 256, 0, stream>>>(xbar_n, W_Q0, W_QT, qt);
    qs_kernel<<<NB * DK, 256, 0, stream>>>(q0, W_QS, qs);
    attn_kernel<<<NB, 32, 0, stream>>>(qt, qs, K_T, K_S, attn_t, attn_s);
    ets_kernel<<<(NB * NN * DC + 255) / 256, 256, 0, stream>>>(V_T, V_S, attn_t, attn_s, etsb);
    z0_kernel<<<(NB * NT * DOUT * NN) / 256, 256, 0, stream>>>(x, W2, z0);

    dim3 grid(NN / 16, NB);
    conv_kernel<<<grid, 384, 0, stream>>>(etsb, dis, z0, x, W_s0, (void*)z1, 0);
    conv_kernel<<<grid, 384, 0, stream>>>(etsb, dis, z1, x, W_s1, d_out, 1);
}
